// CTMCritic_70007966925009
// MI455X (gfx1250) — matrix operand order repaired and checked by reading, not compiled, not run on hardware
//
#include <hip/hip_runtime.h>
#include <stdint.h>

typedef __bf16 bf16;
typedef __attribute__((ext_vector_type(16))) __bf16 v16bf;
typedef __attribute__((ext_vector_type(8)))  float  v8f;

union FragBF { v16bf v; uint4 q[2]; bf16 e[16]; };
union FragF  { v8f  v; float e[8];  };

constexpr int CBATCH = 64;
constexpr int CTRAJ  = 64;
constexpr int CNA    = 8;
constexpr int COBS   = 32;
constexpr int CACTD  = 8;
constexpr int CDIN   = 256;
constexpr int CDM    = 2048;
constexpr int CMEM   = 32;
constexpr int CNS    = 256;
constexpr int CITER  = 16;
constexpr int CKSY   = 32896;              // 256*257/2
constexpr int CRAW   = 4096;               // 2*D_MODEL raw GEMM width
constexpr int CTOK   = CTRAJ + CNA;        // 72 kv tokens
constexpr int CPRE   = CDIN + CDM;         // 2304
constexpr int CSYNB  = (CKSY + 511) / 512; // 65 pair-chunks

__device__ __forceinline__ bf16 f2bf(float x) {
  unsigned u = __builtin_bit_cast(unsigned, x);
  unsigned r = u + 0x7FFFu + ((u >> 16) & 1u);   // round-to-nearest-even
  unsigned short h = (unsigned short)(r >> 16);
  return __builtin_bit_cast(bf16, h);
}
__device__ __forceinline__ float sigm(float x) { return 1.0f / (1.0f + __expf(-x)); }

// ---------------- setup kernels ----------------

// W (K x 4096) f32 row-major  ->  Wt (4096 x K) bf16 column-major (per output col, K contiguous)
__global__ void k_cvt_bf16_t(const float* __restrict__ src, bf16* __restrict__ dst, int Kd) {
  int i = blockIdx.x * blockDim.x + threadIdx.x;       // over Kd*4096
  if (i >= Kd * CRAW) return;
  int k = i / CRAW, n = i % CRAW;                      // coalesced read of src[k][n]
  dst[(size_t)n * Kd + k] = f2bf(src[i]);
}

// sl1_w (MEM,2H,DM): w[m][o][n] -> per-neuron column-major [n][o*32+m] bf16
__global__ void k_sl1_repack(const float* __restrict__ src, bf16* __restrict__ dst) {
  int n = blockIdx.x;
  for (int e = threadIdx.x; e < CMEM * 64; e += blockDim.x) {
    int o = e >> 5, m = e & 31;
    dst[(size_t)n * (CMEM * 64) + e] = f2bf(src[((size_t)(m * 64 + o)) * CDM + n]);
  }
}

// sl2_w (HID,2,DM): w[h][o][n] -> [n][o*32+h] f32
__global__ void k_sl2_repack(const float* __restrict__ src, float* __restrict__ dst) {
  int i = blockIdx.x * blockDim.x + threadIdx.x;
  if (i >= CDM * 64) return;
  int n = i >> 6, e = i & 63, o = e >> 5, h = e & 31;
  dst[n * 64 + o * 32 + h] = src[((size_t)(h * 2 + o)) * CDM + n];
}

__global__ void k_init_tr(const float* __restrict__ st, const float* __restrict__ sat,
                          float* __restrict__ stateT, float* __restrict__ actT) {
  int i = blockIdx.x * blockDim.x + threadIdx.x;
  const int per = CDM * CMEM;
  if (i < CBATCH * per) {
    int rem = i % per;
    stateT[i] = st[rem];
    actT[i]   = sat[rem];
  }
}

// kv row: project token + LayerNorm(256) ; one block per (b, token)
__global__ void k_kv(const float* __restrict__ traj, const float* __restrict__ obs,
                     const float* __restrict__ acts,
                     const float* __restrict__ wtr, const float* __restrict__ btr,
                     const float* __restrict__ wag, const float* __restrict__ bag,
                     const float* __restrict__ lns, const float* __restrict__ lnb,
                     float* __restrict__ kv) {
  int row = blockIdx.x;
  int b = row / CTOK, t = row % CTOK;
  int tid = threadIdx.x;
  __shared__ float vin[64];
  __shared__ float red[256];
  if (t < CTRAJ) {
    if (tid < 32) {
      int a = tid >> 2, c = tid & 3;
      float v;
      if (c < 2) v = traj[((size_t)(b * CTRAJ + t) * CNA + a) * 2 + c];
      else if (t == 0) v = 0.0f;
      else {
        float cur = traj[((size_t)(b * CTRAJ + t)     * CNA + a) * 2 + (c - 2)];
        float prv = traj[((size_t)(b * CTRAJ + t - 1) * CNA + a) * 2 + (c - 2)];
        v = (cur - prv) * 1.25f;                 // 1/VEL_MAX
      }
      vin[tid] = v;
    }
  } else {
    int a = t - CTRAJ;
    if (tid < 40)
      vin[tid] = (tid < COBS) ? obs[(b * CNA + a) * COBS + tid]
                              : acts[(b * CNA + a) * CACTD + (tid - COBS)];
  }
  __syncthreads();
  float dot;
  if (t < CTRAJ) {
    dot = btr[tid];
    #pragma unroll 8
    for (int e = 0; e < 32; ++e) dot += vin[e] * wtr[e * CDIN + tid];
  } else {
    dot = bag[tid];
    for (int e = 0; e < 40; ++e) dot += vin[e] * wag[e * CDIN + tid];
  }
  red[tid] = dot; __syncthreads();
  for (int s = 128; s > 0; s >>= 1) { if (tid < s) red[tid] += red[tid + s]; __syncthreads(); }
  float mu = red[0] * (1.0f / 256.0f);
  __syncthreads();
  float d = dot - mu;
  red[tid] = d * d; __syncthreads();
  for (int s = 128; s > 0; s >>= 1) { if (tid < s) red[tid] += red[tid + s]; __syncthreads(); }
  float var = red[0] * (1.0f / 256.0f);
  kv[(size_t)row * CDIN + tid] = d * rsqrtf(var + 1e-6f) * lns[tid] + lnb[tid];
}

__global__ void k_feat(const float* __restrict__ kv, float* __restrict__ feat) {
  int b = blockIdx.x, j = threadIdx.x;
  float s = 0.f;
  for (int t = 0; t < CTOK; ++t) s += kv[((size_t)(b * CTOK + t)) * CDIN + j];
  feat[b * CDIN + j] = s * (1.0f / CTOK);
}

__global__ void k_pairs(int* __restrict__ pi, int* __restrict__ pj) {
  int i = blockIdx.x, j = threadIdx.x;
  if (j < i) return;
  int base = i * CNS - (i * (i - 1)) / 2;
  int k = base + (j - i);
  pi[k] = i; pj[k] = j;
}

// coef[k][m] = exp(-dp_k * (31-m)) / sqrt(sum+1e-8) * q_w[k]
__global__ void k_coef(const float* __restrict__ dpar, const float* __restrict__ qw,
                       float* __restrict__ coef) {
  int k = blockIdx.x * blockDim.x + threadIdx.x;
  if (k >= CKSY) return;
  float dp = fminf(fmaxf(dpar[k], 0.f), 4.f);
  float ex[CMEM]; float s = 0.f;
  #pragma unroll
  for (int m = 0; m < CMEM; ++m) { ex[m] = __expf(-dp * (float)(CMEM - 1 - m)); s += ex[m]; }
  float scale = qw[k] / sqrtf(s + 1e-8f);
  #pragma unroll
  for (int m = 0; m < CMEM; ++m) coef[(size_t)k * CMEM + m] = ex[m] * scale;
}

// ---------------- per-tick kernels ----------------

__global__ void k_build_pre(const float* __restrict__ feat, const float* __restrict__ actT,
                            bf16* __restrict__ pre) {
  int i = blockIdx.x * blockDim.x + threadIdx.x;
  if (i >= CBATCH * CPRE) return;
  int b = i / CPRE, c = i % CPRE;
  float v = (c < CDIN) ? feat[b * CDIN + c]
                       : actT[((size_t)b * CDM + (c - CDIN)) * CMEM + (CMEM - 1)];
  pre[i] = f2bf(v);
}

// GLU(A @ W + bias) for output cols [c0,c0+64); raw width 4096, GLU pairs j with j+2048.
// Wt is column-major bf16 (4096 x Kd). Block: 256 threads = 8 waves.
__global__ void __launch_bounds__(256)
k_gemm_glu(const bf16* __restrict__ A, int Kd, const bf16* __restrict__ Wt,
           const float* __restrict__ bias, float* __restrict__ G) {
  __shared__ bf16  sA[64 * 32];     // A tile, row-major [row][k]
  __shared__ bf16  sBt[128 * 32];   // B tile, column-major [col][k]
  __shared__ float sR[64 * 128];
  const int tid  = threadIdx.x;
  const int lane = tid & 31;
  const int wv   = tid >> 5;
  const int hsel = wv >> 2;      // 0: cols c0.., 1: cols 2048+c0..
  const int mrow = wv & 3;
  const int c0   = blockIdx.x * 64;

  FragF acc[4];
  #pragma unroll
  for (int nt = 0; nt < 4; ++nt)
    #pragma unroll
    for (int r = 0; r < 8; ++r) acc[nt].e[r] = 0.f;

  const int arow = tid >> 2;
  const int aseg = (tid & 3) * 8;

  for (int k0 = 0; k0 < Kd; k0 += 32) {
    // A tile (64x32 bf16) via async copy to LDS (ASYNCcnt path)
    {
      unsigned ldsoff = (unsigned)(size_t)&sA[arow * 32 + aseg];
      uint64_t gaddr  = (uint64_t)(A + (size_t)arow * Kd + k0 + aseg);
      asm volatile("global_load_async_to_lds_b128 %0, %1, off"
                   :: "v"(ldsoff), "v"(gaddr) : "memory");
    }
    // B tile: 128 cols x 32 k, contiguous 16B chunks both sides
    #pragma unroll
    for (int cidx = 0; cidx < 2; ++cidx) {
      int c = tid * 2 + cidx;                 // 0..511
      int col = c >> 2, seg = c & 3;          // col 0..127, seg 0..3 (8 bf16 per seg)
      int graw = (col < 64) ? (c0 + col) : (2048 + c0 + (col - 64));
      const bf16* srcp = Wt + (size_t)graw * Kd + k0 + seg * 8;
      if (cidx == 0 && k0 + 32 < Kd) __builtin_prefetch(srcp + 32, 0, 0);
      *(uint4*)&sBt[col * 32 + seg * 8] = *(const uint4*)srcp;
    }
    asm volatile("s_wait_asynccnt 0" ::: "memory");
    __syncthreads();

    FragBF fa;   // 16x32 bf16 A fragment: lane<16 K0..15, lane>=16 K16..31, contiguous
    {
      const uint4* pa = (const uint4*)&sA[(mrow * 16 + (lane & 15)) * 32 + ((lane < 16) ? 0 : 16)];
      fa.q[0] = pa[0];
      fa.q[1] = pa[1];     // +8 elements (= +16B)
    }
    #pragma unroll
    for (int nt = 0; nt < 4; ++nt) {
      FragBF fb; // 32x16 bf16 B fragment: lane<16 K0..15, lane>=16 K16..31, contiguous
      const int col = hsel * 64 + nt * 16 + (lane & 15);
      const uint4* pb = (const uint4*)&sBt[col * 32 + ((lane < 16) ? 0 : 16)];
      fb.q[0] = pb[0];
      fb.q[1] = pb[1];
      acc[nt].v = __builtin_amdgcn_wmma_f32_16x16x32_bf16(
          false, fa.v, false, fb.v, (short)0, acc[nt].v, false, false);
    }
    __syncthreads();
  }

  // epilogue: +bias, stage raw, GLU
  #pragma unroll
  for (int nt = 0; nt < 4; ++nt) {
    const int nc = nt * 16 + (lane & 15);
    const float bv = bias[hsel * 2048 + c0 + nc];
    const int rb = mrow * 16 + ((lane >> 4) << 3);   // C/D: vgpr r -> M=r / M=8+r
    #pragma unroll
    for (int r = 0; r < 8; ++r) sR[(rb + r) * 128 + hsel * 64 + nc] = acc[nt].e[r] + bv;
  }
  __syncthreads();
  for (int e2 = tid; e2 < 64 * 64; e2 += 256) {
    int row = e2 >> 6, col = e2 & 63;
    float a = sR[row * 128 + col];
    float g = sR[row * 128 + 64 + col];
    G[(size_t)row * CDM + c0 + col] = a * sigm(g);
  }
}

// LayerNorm over D_MODEL; optional f32 and bf16 outputs
__global__ void k_ln(const float* __restrict__ X, const float* __restrict__ sc,
                     const float* __restrict__ bi, float* outF, bf16* outB) {
  int b = blockIdx.x, tid = threadIdx.x;
  __shared__ float red[256];
  float v[8]; float sum = 0.f;
  #pragma unroll
  for (int i = 0; i < 8; ++i) { v[i] = X[(size_t)b * CDM + tid * 8 + i]; sum += v[i]; }
  red[tid] = sum; __syncthreads();
  for (int s = 128; s > 0; s >>= 1) { if (tid < s) red[tid] += red[tid + s]; __syncthreads(); }
  float mu = red[0] * (1.0f / CDM);
  __syncthreads();
  float sq = 0.f;
  #pragma unroll
  for (int i = 0; i < 8; ++i) { float d = v[i] - mu; sq += d * d; }
  red[tid] = sq; __syncthreads();
  for (int s = 128; s > 0; s >>= 1) { if (tid < s) red[tid] += red[tid + s]; __syncthreads(); }
  float rs = rsqrtf(red[0] * (1.0f / CDM) + 1e-6f);
  #pragma unroll
  for (int i = 0; i < 8; ++i) {
    int j = tid * 8 + i;
    float y = (v[i] - mu) * rs * sc[j] + bi[j];
    if (outF) outF[(size_t)b * CDM + j] = y;
    if (outB) outB[(size_t)b * CDM + j] = f2bf(y);
  }
}

__global__ void k_shift(float* __restrict__ tr, const float* __restrict__ s) {
  int i = blockIdx.x * blockDim.x + threadIdx.x;
  if (i < CBATCH * CDM) {
    float* p = tr + (size_t)i * CMEM;
    #pragma unroll
    for (int m = 0; m < CMEM - 1; ++m) p[m] = p[m + 1];
    p[CMEM - 1] = s[i];
  }
}

// per-neuron: (64x32)@(32x64) WMMA (single K=32 step) + GLU + (32->2) + GLU; shift act_trace
__global__ void __launch_bounds__(128)
k_neuron(const float* __restrict__ state, const bf16* __restrict__ w1,
         const float* __restrict__ b1, const float* __restrict__ w2,
         const float* __restrict__ b2, float* __restrict__ actT) {
  __shared__ bf16  sA[64 * 32];     // [batch][m]
  __shared__ bf16  sB[64 * 32];     // column-major [o][m]
  __shared__ float sR[64 * 64];
  const int n = blockIdx.x;
  const int tid = threadIdx.x;
  const int lane = tid & 31;
  const int wv = tid >> 5;
  {
    // convert 16 f32 -> 16 bf16 in registers, 2x b128 store to LDS
    int b = tid >> 1, ms = (tid & 1) * 16;
    const float4* src = (const float4*)(state + ((size_t)b * CDM + n) * CMEM + ms);
    FragBF buf;
    #pragma unroll
    for (int c4 = 0; c4 < 4; ++c4) {
      float4 f = src[c4];
      buf.e[c4 * 4 + 0] = f2bf(f.x);
      buf.e[c4 * 4 + 1] = f2bf(f.y);
      buf.e[c4 * 4 + 2] = f2bf(f.z);
      buf.e[c4 * 4 + 3] = f2bf(f.w);
    }
    uint4* dstp = (uint4*)&sA[b * 32 + ms];
    dstp[0] = buf.q[0];
    dstp[1] = buf.q[1];
  }
  {
    const bf16* src = w1 + (size_t)n * (CMEM * 64) + tid * 16;   // already [o][m]
    uint4* dstp = (uint4*)&sB[tid * 16];
    dstp[0] = ((const uint4*)src)[0];
    dstp[1] = ((const uint4*)src)[1];
  }
  __syncthreads();
  FragBF fa;
  {
    const uint4* pa = (const uint4*)&sA[(wv * 16 + (lane & 15)) * 32 + ((lane < 16) ? 0 : 16)];
    fa.q[0] = pa[0];
    fa.q[1] = pa[1];
  }
  #pragma unroll
  for (int nt = 0; nt < 4; ++nt) {
    FragBF fb;
    const int c = nt * 16 + (lane & 15);
    const uint4* pb = (const uint4*)&sB[c * 32 + ((lane < 16) ? 0 : 16)];
    fb.q[0] = pb[0];
    fb.q[1] = pb[1];
    FragF acc;
    #pragma unroll
    for (int r = 0; r < 8; ++r) acc.e[r] = 0.f;
    acc.v = __builtin_amdgcn_wmma_f32_16x16x32_bf16(
        false, fa.v, false, fb.v, (short)0, acc.v, false, false);
    const float bv = b1[(size_t)n * 64 + c];
    const int rb = wv * 16 + ((lane >> 4) << 3);
    #pragma unroll
    for (int r = 0; r < 8; ++r) sR[(rb + r) * 64 + c] = acc.e[r] + bv;
  }
  __syncthreads();
  if (tid < 64) {
    const int b = tid;
    float tp[32];
    #pragma unroll
    for (int h = 0; h < 32; ++h) tp[h] = sR[b * 64 + h] * sigm(sR[b * 64 + 32 + h]);
    float d0 = b2[n * 2 + 0], d1 = b2[n * 2 + 1];
    const float* wvp = w2 + (size_t)n * 64;
    #pragma unroll
    for (int h = 0; h < 32; ++h) { d0 += tp[h] * wvp[h]; d1 += tp[h] * wvp[32 + h]; }
    float na = d0 * sigm(d1);
    float* at = actT + ((size_t)b * CDM + n) * CMEM;
    #pragma unroll
    for (int m = 0; m < CMEM - 1; ++m) at[m] = at[m + 1];
    at[CMEM - 1] = na;
  }
}

// synch partial sums: block (chunk cx, batch b) handles 512 pairs
__global__ void k_synch(const float* __restrict__ actT, const float* __restrict__ coef,
                        const int* __restrict__ pi, const int* __restrict__ pj,
                        float* __restrict__ part) {
  const int b = blockIdx.y, cx = blockIdx.x, tid = threadIdx.x;
  __shared__ float sS[CMEM * CNS];     // 32 KB
  __shared__ float red[256];
  for (int e = tid; e < CMEM * CNS; e += 256) {
    int m = e >> 8, j = e & 255;
    sS[e] = actT[((size_t)b * CDM + (CDM - CNS) + j) * CMEM + m];
  }
  __syncthreads();
  float local = 0.f;
  #pragma unroll
  for (int p = 0; p < 2; ++p) {
    int k = cx * 512 + p * 256 + tid;
    if (k < CKSY) {
      int i = pi[k], j = pj[k];
      const float* ck = coef + (size_t)k * CMEM;
      float a = 0.f;
      #pragma unroll
      for (int m = 0; m < CMEM; ++m) a += sS[m * CNS + i] * sS[m * CNS + j] * ck[m];
      local += a;
    }
  }
  red[tid] = local; __syncthreads();
  for (int s = 128; s > 0; s >>= 1) { if (tid < s) red[tid] += red[tid + s]; __syncthreads(); }
  if (tid == 0) part[b * CSYNB + cx] = red[0];
}

__global__ void k_qred(const float* __restrict__ part, const float* __restrict__ qb,
                       float* __restrict__ out, int tick) {
  int b = threadIdx.x;
  if (b < CBATCH) {
    float s = qb[0];
    for (int c = 0; c < CSYNB; ++c) s += part[b * CSYNB + c];
    out[b * CITER + tick] = s;
  }
}

// ---------------- host ----------------

extern "C" void kernel_launch(void* const* d_in, const int* in_sizes, int n_in,
                              void* d_out, int out_size, void* d_ws, size_t ws_size,
                              hipStream_t stream) {
  (void)in_sizes; (void)n_in; (void)out_size; (void)ws_size;
  const float* traj   = (const float*)d_in[0];
  const float* obs    = (const float*)d_in[1];
  const float* acts   = (const float*)d_in[2];
  const float* p_trw  = (const float*)d_in[3];
  const float* p_trb  = (const float*)d_in[4];
  const float* p_agw  = (const float*)d_in[5];
  const float* p_agb  = (const float*)d_in[6];
  const float* p_kvs  = (const float*)d_in[7];
  const float* p_kvb  = (const float*)d_in[8];
  const float* p_s1w  = (const float*)d_in[9];
  const float* p_s1b  = (const float*)d_in[10];
  const float* p_l1s  = (const float*)d_in[11];
  const float* p_l1b  = (const float*)d_in[12];
  const float* p_s2w  = (const float*)d_in[13];
  const float* p_s2b  = (const float*)d_in[14];
  const float* p_l2s  = (const float*)d_in[15];
  const float* p_l2b  = (const float*)d_in[16];
  const float* p_sl1w = (const float*)d_in[17];
  const float* p_sl1b = (const float*)d_in[18];
  const float* p_sl2w = (const float*)d_in[19];
  const float* p_sl2b = (const float*)d_in[20];
  const float* p_qw   = (const float*)d_in[21];
  const float* p_qb   = (const float*)d_in[22];
  const float* p_st   = (const float*)d_in[23];
  const float* p_sat  = (const float*)d_in[24];
  const float* p_dp   = (const float*)d_in[25];

  char* wsb = (char*)d_ws;
  size_t off = 0;
  auto alloc = [&](size_t bytes) -> void* {
    void* p = wsb + off;
    off = (off + bytes + 255) & ~(size_t)255;
    return p;
  };
  bf16*  syn1t  = (bf16*) alloc((size_t)CPRE * CRAW * 2);   // [4096][2304]
  bf16*  syn2t  = (bf16*) alloc((size_t)CDM  * CRAW * 2);   // [4096][2048]
  bf16*  sl1bf  = (bf16*) alloc((size_t)CDM * CMEM * 64 * 2);
  float* sl2r   = (float*)alloc((size_t)CDM * 64 * 4);
  float* stateT = (float*)alloc((size_t)CBATCH * CDM * CMEM * 4);
  float* actT   = (float*)alloc((size_t)CBATCH * CDM * CMEM * 4);
  float* kv     = (float*)alloc((size_t)CBATCH * CTOK * CDIN * 4);
  float* feat   = (float*)alloc((size_t)CBATCH * CDIN * 4);
  bf16*  prebf  = (bf16*) alloc((size_t)CBATCH * CPRE * 2);
  float* g1     = (float*)alloc((size_t)CBATCH * CDM * 4);
  bf16*  s1bf   = (bf16*) alloc((size_t)CBATCH * CDM * 2);
  float* g2     = (float*)alloc((size_t)CBATCH * CDM * 4);
  float* s2     = (float*)alloc((size_t)CBATCH * CDM * 4);
  float* coef   = (float*)alloc((size_t)CKSY * CMEM * 4);
  int*   pi     = (int*)  alloc((size_t)CKSY * 4);
  int*   pj     = (int*)  alloc((size_t)CKSY * 4);
  float* qpart  = (float*)alloc((size_t)CBATCH * CSYNB * 4);
  float* outF   = (float*)d_out;

  // one-time setup (per call; deterministic)
  k_cvt_bf16_t<<<(CPRE * CRAW + 255) / 256, 256, 0, stream>>>(p_s1w, syn1t, CPRE);
  k_cvt_bf16_t<<<(CDM  * CRAW + 255) / 256, 256, 0, stream>>>(p_s2w, syn2t, CDM);
  k_sl1_repack<<<CDM, 128, 0, stream>>>(p_sl1w, sl1bf);
  k_sl2_repack<<<(CDM * 64 + 255) / 256, 256, 0, stream>>>(p_sl2w, sl2r);
  k_init_tr<<<(CBATCH * CDM * CMEM + 255) / 256, 256, 0, stream>>>(p_st, p_sat, stateT, actT);
  k_kv<<<CBATCH * CTOK, 256, 0, stream>>>(traj, obs, acts, p_trw, p_trb, p_agw, p_agb,
                                          p_kvs, p_kvb, kv);
  k_feat<<<CBATCH, 256, 0, stream>>>(kv, feat);
  k_pairs<<<CNS, CNS, 0, stream>>>(pi, pj);
  k_coef<<<(CKSY + 255) / 256, 256, 0, stream>>>(p_dp, p_qw, coef);

  // 16 sequential ticks
  for (int t = 0; t < CITER; ++t) {
    k_build_pre<<<(CBATCH * CPRE + 255) / 256, 256, 0, stream>>>(feat, actT, prebf);
    k_gemm_glu<<<CDM / 64, 256, 0, stream>>>(prebf, CPRE, syn1t, p_s1b, g1);
    k_ln<<<CBATCH, 256, 0, stream>>>(g1, p_l1s, p_l1b, nullptr, s1bf);
    k_gemm_glu<<<CDM / 64, 256, 0, stream>>>(s1bf, CDM, syn2t, p_s2b, g2);
    k_ln<<<CBATCH, 256, 0, stream>>>(g2, p_l2s, p_l2b, s2, nullptr);
    k_shift<<<(CBATCH * CDM + 255) / 256, 256, 0, stream>>>(stateT, s2);
    k_neuron<<<CDM, 128, 0, stream>>>(stateT, sl1bf, p_sl1b, sl2r, p_sl2b, actT);
    k_synch<<<dim3(CSYNB, CBATCH), 256, 0, stream>>>(actT, coef, pi, pj, qpart);
    k_qred<<<1, 64, 0, stream>>>(qpart, p_qb, outF, t);
  }
}
